// DualAttenion_33346126086788
// MI455X (gfx1250) — matrix operand order repaired and checked by reading, not compiled, not run on hardware
//
#include <hip/hip_runtime.h>
#include <hip/hip_bf16.h>
#include <math.h>

// ---------------------------------------------------------------------------
// DualAttention for MI455X (gfx1250), bf16 WMMA + f32 accumulation.
// B=32, nvars=32, H=64, C=256, nh=8, hd=32, dp_rank=16, d_ff=512.
// Row space: M = B*nvars*H = 65536 rows of C=256 channels.
// GEMMs: double-buffered LDS; weight/bf16 tiles moved by the Tensor Data
// Mover (TENSOR_LOAD_TO_LDS + s_wait_tensorcnt) when the builtin exists,
// register-staged copies otherwise.
// ---------------------------------------------------------------------------

typedef unsigned short u16;
typedef unsigned int   u32;
typedef __attribute__((ext_vector_type(16))) __bf16 v16bf;
typedef __attribute__((ext_vector_type(8)))  float  v8f;
typedef __attribute__((ext_vector_type(4)))  u32    u32x4;
typedef __attribute__((ext_vector_type(4)))  int    i32x4;
typedef __attribute__((ext_vector_type(8)))  int    i32x8;

#if defined(__has_builtin)
#if __has_builtin(__builtin_amdgcn_tensor_load_to_lds) && __has_builtin(__builtin_amdgcn_s_wait_tensorcnt)
#define HAVE_TDM 1
#endif
#endif
#ifndef HAVE_TDM
#define HAVE_TDM 0
#endif

#define M_ROWS 65536
#define SQRT_HD 5.656854249492380f   // sqrt(32)
#define SQRT_H  8.0f                 // sqrt(64)

static __device__ __forceinline__ u16 f2bf(float f) {
    u32 u = __float_as_uint(f);
    u32 r = u + 0x7FFFu + ((u >> 16) & 1u);   // round-to-nearest-even
    return (u16)(r >> 16);
}

// A-fragment (16xK=32 bf16): lane m=l&15, hi=l>>4; elems 0..15 = K hi*16 .. hi*16+15
// (same K order as lds_fragB).
static __device__ __forceinline__ v16bf lds_fragA(const u16* tile, int row0, int pitch, int k0, int lane) {
    const int m = lane & 15, hi = lane >> 4;
    const u16* p = tile + (row0 + m) * pitch + k0;
    union { v16bf v; uint4 q[2]; } f;
    f.q[0] = *(const uint4*)(p + hi * 16);
    f.q[1] = *(const uint4*)(p + hi * 16 + 8);
    return f.v;
}

// B-fragment from N-major (Bt[n][k]) storage: lane n=l&15, hi=l>>4;
// elems 0..15 = K hi*16 .. hi*16+15 (contiguous K order, same as lds_fragA).
static __device__ __forceinline__ v16bf lds_fragB(const u16* tileT, int n0, int pitch, int k0, int lane) {
    const int n = lane & 15, hi = lane >> 4;
    const u16* p = tileT + (n0 + n) * pitch + k0 + hi * 16;
    union { v16bf v; uint4 q[2]; } f;
    f.q[0] = *(const uint4*)(p);
    f.q[1] = *(const uint4*)(p + 8);
    return f.v;
}

static __device__ __forceinline__ v8f wmma_bf16(v16bf a, v16bf b, v8f c) {
    return __builtin_amdgcn_wmma_f32_16x16x32_bf16(false, a, false, b, (short)0, c, false, false);
}

// ---------------------------------------------------------------------------
// TDM: DMA a (tile_h x 32) bf16 tile (row stride = stride_elems) from global
// memory into contiguous LDS at lds_dst. D# bitfields per CDNA5 ISA §8.
// Issue from one full wave; completion via s_wait_tensorcnt 0.
// ---------------------------------------------------------------------------
static __device__ __forceinline__ void tdm_fill_tile(const u16* gsrc, u16* lds_dst,
                                                     u32 tile_h, u32 stride_elems) {
#if HAVE_TDM
    unsigned long long ga = (unsigned long long)(size_t)gsrc;
    u32 lds_off = (u32)(size_t)lds_dst;      // low 32 bits of LDS generic pointer = LDS byte offset
    u32x4 g0;
    g0.x = 1u;                                // count=1, user descriptor, no gather
    g0.y = lds_off;                           // D#.lds_addr
    g0.z = (u32)ga;                           // D#.global_addr[31:0]
    g0.w = (u32)(ga >> 32) | (2u << 30);      // D#.global_addr[56:32] | type=2 ("image")
    i32x8 g1;
    g1[0] = 0x00010000;                       // workgroup_mask=0, data_size=1 (2 bytes)
    g1[1] = (int)((stride_elems & 0xFFFFu) << 16);                     // tensor_dim0[15:0]
    g1[2] = (int)(((stride_elems >> 16) & 0xFFFFu) | (0xFFFFu << 16)); // dim0[31:16] | tensor_dim1[15:0]=big
    g1[3] = (int)(32u << 16);                 // tensor_dim1[31:16]=0 | tile_dim0=32
    g1[4] = (int)(tile_h & 0xFFFFu);          // tile_dim1 | tile_dim2=0
    g1[5] = (int)stride_elems;                // tensor_dim0_stride[31:0]
    g1[6] = 0;                                // stride[47:32] | tensor_dim1_stride lo (unused, 2D)
    g1[7] = 0;
    i32x4 z4 = {0, 0, 0, 0};
#if __clang_major__ >= 23
    i32x8 z8 = {0, 0, 0, 0, 0, 0, 0, 0};
    __builtin_amdgcn_tensor_load_to_lds(g0, g1, z4, z4, z8, 0);
#else
    __builtin_amdgcn_tensor_load_to_lds(g0, g1, z4, z4, 0);
#endif
#else
    (void)gsrc; (void)lds_dst; (void)tile_h; (void)stride_elems;
#endif
}

static __device__ __forceinline__ void tdm_wait() {
#if HAVE_TDM
    __builtin_amdgcn_s_wait_tensorcnt((short)0);
#endif
}

// ---------------------------------------------------------------------------
// Weight prep: fp32 (KxN) -> bf16 N-major (NxK)
// ---------------------------------------------------------------------------
__global__ void k_convert_transpose(const float* __restrict__ in, u16* __restrict__ out, int K, int N) {
    int idx = blockIdx.x * 256 + threadIdx.x;
    if (idx >= K * N) return;
    int n = idx / K, k = idx - n * K;
    out[idx] = f2bf(in[k * N + n]);
}

// small operands: w_dpk^T, w_dpv^T (16x32), ema64 (64x64 A-major), ema16 padded (16x32)
__global__ void k_prep_small(const float* __restrict__ w_dpk, const float* __restrict__ w_dpv,
                             const float* __restrict__ ema,
                             u16* __restrict__ wkT, u16* __restrict__ wvT,
                             u16* __restrict__ em64, u16* __restrict__ em16) {
    int i = blockIdx.x * 256 + threadIdx.x;
    if (i < 512) {                       // wkT[c*32+d] = w_dpk[d][c], w_dpk is (32x16)
        int c = i >> 5, d = i & 31;
        wkT[i] = f2bf(w_dpk[d * 16 + c]);
    } else if (i < 1024) {
        int j = i - 512, c = j >> 5, d = j & 31;
        wvT[j] = f2bf(w_dpv[d * 16 + c]);
    } else if (i < 1024 + 4096) {        // ema is 64x64 row-major, used as A directly
        int j = i - 1024;
        em64[j] = f2bf(ema[j]);
    } else if (i < 1024 + 4096 + 512) {  // ema[:16,:16], K padded to 32 with zeros
        int j = i - 5120, g = j >> 5, a = j & 31;
        em16[j] = (a < 16) ? f2bf(ema[g * 64 + a]) : (u16)0;
    }
}

// ---------------------------------------------------------------------------
// GEMM: out_bf16[M x N] = f( affine(A_f32[M x K]) @ W[K x N] + bias )
// W passed as bf16 N-major (Wt[N][K]). Block tile 128x128, BK=32, 8 waves,
// wave tile 32x64 -> 8 v_wmma per K-step. Double-buffered LDS; B tile via TDM.
// ---------------------------------------------------------------------------
__global__ __launch_bounds__(256) void k_gemm_a32(
        const float* __restrict__ A, const u16* __restrict__ Wt,
        const float* __restrict__ bias, const float* __restrict__ scsh,  // scale[0..K), shift[K..2K) or null
        u16* __restrict__ out, int K, int N, int gelu_flag)
{
    __shared__ __align__(16) u16 As[2][128 * 32];
    __shared__ __align__(16) u16 Bs[2][128 * 32];
    const int tid = threadIdx.x, lane = tid & 31, wave = tid >> 5;
    const int waveM = wave >> 1, waveN = wave & 1;
    const int m0 = blockIdx.x * 128, n0 = blockIdx.y * 128;
    const v8f vz = {0.f, 0.f, 0.f, 0.f, 0.f, 0.f, 0.f, 0.f};
    v8f acc[2][4];
#pragma unroll
    for (int mi = 0; mi < 2; ++mi)
#pragma unroll
        for (int ni = 0; ni < 4; ++ni) acc[mi][ni] = vz;

    const int r = tid >> 1, h = tid & 1;
    const int steps = K / 32;

    float areg[16];
    uint4 br0, br1;

    // --- tile fetch helpers ---
    auto fetchA = [&](int k0) {       // global fp32 -> (BN affine) -> regs
        const float* ap = A + (size_t)(m0 + r) * K + (k0 + h * 16);
        union { float4 f4[4]; float s[16]; } Av;
        Av.f4[0] = ((const float4*)ap)[0];
        Av.f4[1] = ((const float4*)ap)[1];
        Av.f4[2] = ((const float4*)ap)[2];
        Av.f4[3] = ((const float4*)ap)[3];
        if (scsh) {
            int cb = k0 + h * 16;
#pragma unroll
            for (int i = 0; i < 16; ++i) Av.s[i] = Av.s[i] * scsh[cb + i] + scsh[K + cb + i];
        }
#pragma unroll
        for (int i = 0; i < 16; ++i) areg[i] = Av.s[i];
    };
    auto commitA = [&](int buf) {     // regs -> bf16 -> LDS
        union { uint4 q[2]; u16 u[16]; } U;
#pragma unroll
        for (int i = 0; i < 16; ++i) U.u[i] = f2bf(areg[i]);
        *(uint4*)(As[buf] + r * 32 + h * 16)     = U.q[0];
        *(uint4*)(As[buf] + r * 32 + h * 16 + 8) = U.q[1];
    };
    auto fillB = [&](int k0, int buf) {
#if HAVE_TDM
        if (wave == 0) tdm_fill_tile(Wt + (size_t)n0 * K + k0, Bs[buf], 128, (u32)K);
#else
        const u16* bp = Wt + (size_t)(n0 + r) * K + (k0 + h * 16);
        br0 = *(const uint4*)(bp);
        br1 = *(const uint4*)(bp + 8);
        *(uint4*)(Bs[buf] + r * 32 + h * 16)     = br0;
        *(uint4*)(Bs[buf] + r * 32 + h * 16 + 8) = br1;
#endif
    };

    // --- prologue: tile 0 ---
    fillB(0, 0);
    fetchA(0);
    commitA(0);
#if HAVE_TDM
    if (wave == 0) tdm_wait();
#endif
    __syncthreads();

    for (int i = 0; i < steps; ++i) {
        const int cur = i & 1, nxt = cur ^ 1;
        const bool more = (i + 1) < steps;
        if (more) {                  // prefetch next tile (TDM async / global loads)
            fillB((i + 1) * 32, nxt);
            fetchA((i + 1) * 32);
        }
        v16bf af[2], bf[4];
#pragma unroll
        for (int mi = 0; mi < 2; ++mi) af[mi] = lds_fragA(As[cur], waveM * 32 + mi * 16, 32, 0, lane);
#pragma unroll
        for (int ni = 0; ni < 4; ++ni) bf[ni] = lds_fragB(Bs[cur], waveN * 64 + ni * 16, 32, 0, lane);
#pragma unroll
        for (int mi = 0; mi < 2; ++mi)
#pragma unroll
            for (int ni = 0; ni < 4; ++ni) acc[mi][ni] = wmma_bf16(af[mi], bf[ni], acc[mi][ni]);
        if (more) commitA(nxt);
#if HAVE_TDM
        if (more && wave == 0) tdm_wait();
#endif
        __syncthreads();
    }

    const int n = lane & 15, hi = lane >> 4;
#pragma unroll
    for (int mi = 0; mi < 2; ++mi)
#pragma unroll
        for (int ni = 0; ni < 4; ++ni) {
            int nG = n0 + waveN * 64 + ni * 16 + n;
            float bv = bias[nG];
#pragma unroll
            for (int j = 0; j < 8; ++j) {
                int mG = m0 + waveM * 32 + mi * 16 + j + 8 * hi;
                float val = acc[mi][ni][j] + bv;
                if (gelu_flag) val = 0.5f * val * (1.0f + erff(val * 0.70710678118654752f));
                out[(size_t)mG * N + nG] = f2bf(val);
            }
        }
}

// out_f32[M x 256] = h1 @ W2a + h2 @ W2b + b2a + b2b + src   (K = 512 each)
// All operands bf16 -> both tiles TDM-copied; 32 unified K-steps (2 passes x 16).
__global__ __launch_bounds__(256) void k_gemm_ffout(
        const u16* __restrict__ A1, const u16* __restrict__ Wt1,
        const u16* __restrict__ A2, const u16* __restrict__ Wt2,
        const float* __restrict__ b1, const float* __restrict__ b2,
        const float* __restrict__ src, float* __restrict__ out)
{
    __shared__ __align__(16) u16 As[2][128 * 32];
    __shared__ __align__(16) u16 Bs[2][128 * 32];
    const int tid = threadIdx.x, lane = tid & 31, wave = tid >> 5;
    const int waveM = wave >> 1, waveN = wave & 1;
    const int m0 = blockIdx.x * 128, n0 = blockIdx.y * 128;
    const v8f vz = {0.f, 0.f, 0.f, 0.f, 0.f, 0.f, 0.f, 0.f};
    v8f acc[2][4];
#pragma unroll
    for (int mi = 0; mi < 2; ++mi)
#pragma unroll
        for (int ni = 0; ni < 4; ++ni) acc[mi][ni] = vz;

    const int r = tid >> 1, h = tid & 1;
    uint4 ar0, ar1, br0, br1;

    auto fill = [&](int t, int buf) {           // step t in [0,32): pass=t>>4, k0=(t&15)*32
        const u16* Ap = (t < 16) ? A1 : A2;
        const u16* Wp = (t < 16) ? Wt1 : Wt2;
        const int k0 = (t & 15) * 32;
#if HAVE_TDM
        if (wave == 0) {
            tdm_fill_tile(Ap + (size_t)m0 * 512 + k0, As[buf], 128, 512u);
            tdm_fill_tile(Wp + (size_t)n0 * 512 + k0, Bs[buf], 128, 512u);
        }
#else
        const u16* ap = Ap + (size_t)(m0 + r) * 512 + (k0 + h * 16);
        ar0 = *(const uint4*)(ap);
        ar1 = *(const uint4*)(ap + 8);
        const u16* bp = Wp + (size_t)(n0 + r) * 512 + (k0 + h * 16);
        br0 = *(const uint4*)(bp);
        br1 = *(const uint4*)(bp + 8);
#endif
    };
    auto commit = [&](int buf) {
#if !HAVE_TDM
        *(uint4*)(As[buf] + r * 32 + h * 16)     = ar0;
        *(uint4*)(As[buf] + r * 32 + h * 16 + 8) = ar1;
        *(uint4*)(Bs[buf] + r * 32 + h * 16)     = br0;
        *(uint4*)(Bs[buf] + r * 32 + h * 16 + 8) = br1;
#else
        (void)buf;
#endif
    };

    fill(0, 0);
    commit(0);
#if HAVE_TDM
    if (wave == 0) tdm_wait();
#endif
    __syncthreads();

    for (int t = 0; t < 32; ++t) {
        const int cur = t & 1, nxt = cur ^ 1;
        const bool more = (t + 1) < 32;
        if (more) fill(t + 1, nxt);
        v16bf af[2], bf[4];
#pragma unroll
        for (int mi = 0; mi < 2; ++mi) af[mi] = lds_fragA(As[cur], waveM * 32 + mi * 16, 32, 0, lane);
#pragma unroll
        for (int ni = 0; ni < 4; ++ni) bf[ni] = lds_fragB(Bs[cur], waveN * 64 + ni * 16, 32, 0, lane);
#pragma unroll
        for (int mi = 0; mi < 2; ++mi)
#pragma unroll
            for (int ni = 0; ni < 4; ++ni) acc[mi][ni] = wmma_bf16(af[mi], bf[ni], acc[mi][ni]);
        if (more) commit(nxt);
#if HAVE_TDM
        if (more && wave == 0) tdm_wait();
#endif
        __syncthreads();
    }

    const int n = lane & 15, hi = lane >> 4;
#pragma unroll
    for (int mi = 0; mi < 2; ++mi)
#pragma unroll
        for (int ni = 0; ni < 4; ++ni) {
            int nG = n0 + waveN * 64 + ni * 16 + n;
            float bv = b1[nG] + b2[nG];
#pragma unroll
            for (int j = 0; j < 8; ++j) {
                int mG = m0 + waveM * 32 + mi * 16 + j + 8 * hi;
                out[(size_t)mG * 256 + nG] = acc[mi][ni][j] + bv + src[(size_t)mG * 256 + nG];
            }
        }
}

// ---------------------------------------------------------------------------
// Attention: one (b, nvar, head) per block; 4 waves; all matrix products WMMA.
// ---------------------------------------------------------------------------
__global__ __launch_bounds__(128) void k_attn(
        const u16* __restrict__ qkvB,
        const u16* __restrict__ wkT_g, const u16* __restrict__ wvT_g,
        const u16* __restrict__ em64_g, const u16* __restrict__ em16_g,
        const float* __restrict__ b_dpk, const float* __restrict__ b_dpv,
        float* __restrict__ out_tok, float* __restrict__ out_hid)
{
    __shared__ __align__(16) u16 qS[64 * 32], qT[32 * 64], kS[64 * 32], kT[32 * 64];
    __shared__ __align__(16) u16 vS[64 * 32], vT[32 * 64];
    __shared__ __align__(16) u16 eqS[64 * 32], pT[16 * 64], kdpT[32 * 32], vdpT[32 * 32];
    __shared__ __align__(16) u16 ekdpS[16 * 32], atokS[64 * 32], ahidS[32 * 32];
    __shared__ __align__(16) u16 wkT[16 * 32], wvT[16 * 32], em64[64 * 64], em16[16 * 32];
    __shared__ __align__(16) float fS[1024];

    const int tid = threadIdx.x, lane = tid & 31, wave = tid >> 5;
    const int blk = blockIdx.x, bn = blk >> 3, head = blk & 7;
    const int n = lane & 15, hi = lane >> 4;
    const v8f vz = {0.f, 0.f, 0.f, 0.f, 0.f, 0.f, 0.f, 0.f};

    // ---- load q/k/v tiles (+ transposes) and constants ----
    {
        const int r = tid >> 1, h = tid & 1;
        const u16* row = qkvB + (size_t)(bn * 64 + r) * 768;
#pragma unroll
        for (int w = 0; w < 3; ++w) {
            const u16* p = row + w * 256 + head * 32 + h * 16;
            union { uint4 q[2]; u16 u[16]; } U;
            U.q[0] = *(const uint4*)(p);
            U.q[1] = *(const uint4*)(p + 8);
            u16* dst = (w == 0) ? qS : (w == 1) ? kS : vS;
            *(uint4*)(dst + r * 32 + h * 16)     = U.q[0];
            *(uint4*)(dst + r * 32 + h * 16 + 8) = U.q[1];
            u16* dT = (w == 0) ? qT : (w == 1) ? kT : vT;
#pragma unroll
            for (int i = 0; i < 16; ++i) dT[(h * 16 + i) * 64 + r] = U.u[i];
        }
        for (int i = tid; i < 512; i += 128) { ((u32*)kdpT)[i] = 0u; ((u32*)vdpT)[i] = 0u; }
        for (int i = tid; i < 256; i += 128) {
            ((u32*)wkT)[i]  = ((const u32*)wkT_g)[i];
            ((u32*)wvT)[i]  = ((const u32*)wvT_g)[i];
            ((u32*)em16)[i] = ((const u32*)em16_g)[i];
        }
        for (int i = tid; i < 2048; i += 128) ((u32*)em64)[i] = ((const u32*)em64_g)[i];
    }
    __syncthreads();

    // ---- dynamic projection for K (pass 0) and V (pass 1) ----
    for (int pass = 0; pass < 2; ++pass) {
        const u16* xS = pass ? vS : kS;
        const u16* xT = pass ? vT : kT;
        const u16* wT = pass ? wvT : wkT;
        const float* bb = pass ? b_dpv : b_dpk;
        u16* dpT = pass ? vdpT : kdpT;
        { // scores(64x16) = x @ w_dp
            v16bf a = lds_fragA(xS, wave * 16, 32, 0, lane);
            v16bf b = lds_fragB(wT, 0, 32, 0, lane);
            v8f d = wmma_bf16(a, b, vz);
#pragma unroll
            for (int j = 0; j < 8; ++j) fS[(wave * 16 + j + 8 * hi) * 16 + n] = d[j];
        }
        __syncthreads();
        if (tid < 64) { // row softmax over 16
            float sv[16], mx = -1e30f;
#pragma unroll
            for (int c = 0; c < 16; ++c) { sv[c] = fS[tid * 16 + c] + bb[c]; mx = fmaxf(mx, sv[c]); }
            float s = 0.f;
#pragma unroll
            for (int c = 0; c < 16; ++c) { sv[c] = __expf(sv[c] - mx); s += sv[c]; }
            float inv = 1.0f / s;
#pragma unroll
            for (int c = 0; c < 16; ++c) pT[c * 64 + tid] = f2bf(sv[c] * inv);
        }
        __syncthreads();
        if (wave < 2) { // x_dp(16x32) = pT @ x, stored transposed as dpT[d][c]
            const int n0 = wave * 16;
            v8f acc = vz;
#pragma unroll
            for (int kk = 0; kk < 64; kk += 32) {
                v16bf a = lds_fragA(pT, 0, 64, kk, lane);
                v16bf b = lds_fragB(xT, n0, 64, kk, lane);
                acc = wmma_bf16(a, b, acc);
            }
#pragma unroll
            for (int j = 0; j < 8; ++j) dpT[(n0 + n) * 32 + (j + 8 * hi)] = f2bf(acc[j]);
        }
        __syncthreads();
    }

    // ---- eq = ema64 @ q ----
    {
        v8f a0 = vz, a1 = vz;
#pragma unroll
        for (int kk = 0; kk < 64; kk += 32) {
            v16bf a  = lds_fragA(em64, wave * 16, 64, kk, lane);
            v16bf b0 = lds_fragB(qT, 0, 64, kk, lane);
            v16bf b1 = lds_fragB(qT, 16, 64, kk, lane);
            a0 = wmma_bf16(a, b0, a0);
            a1 = wmma_bf16(a, b1, a1);
        }
#pragma unroll
        for (int j = 0; j < 8; ++j) {
            int m = wave * 16 + j + 8 * hi;
            eqS[m * 32 + n]      = f2bf(a0[j]);
            eqS[m * 32 + 16 + n] = f2bf(a1[j]);
        }
    }
    __syncthreads();
    // ---- ekdp = ema16(pad) @ k_dp ----
    if (wave < 2) {
        const int n0 = wave * 16;
        v16bf a = lds_fragA(em16, 0, 32, 0, lane);
        v16bf b = lds_fragB(kdpT, n0, 32, 0, lane);
        v8f d = wmma_bf16(a, b, vz);
#pragma unroll
        for (int j = 0; j < 8; ++j) ekdpS[(j + 8 * hi) * 32 + n0 + n] = f2bf(d[j]);
    }
    __syncthreads();
    // ---- s_tok = eq @ ekdp^T * sqrt(hd); softmax -> a_tok (K padded to 32) ----
    {
        v16bf a = lds_fragA(eqS, wave * 16, 32, 0, lane);
        v16bf b = lds_fragB(ekdpS, 0, 32, 0, lane);
        v8f d = wmma_bf16(a, b, vz);
#pragma unroll
        for (int j = 0; j < 8; ++j) fS[(wave * 16 + j + 8 * hi) * 16 + n] = d[j] * SQRT_HD;
    }
    __syncthreads();
    if (tid < 64) {
        float sv[16], mx = -1e30f;
#pragma unroll
        for (int c = 0; c < 16; ++c) { sv[c] = fS[tid * 16 + c]; mx = fmaxf(mx, sv[c]); }
        float s = 0.f;
#pragma unroll
        for (int c = 0; c < 16; ++c) { sv[c] = __expf(sv[c] - mx); s += sv[c]; }
        float inv = 1.0f / s;
#pragma unroll
        for (int c = 0; c < 16; ++c) {
            atokS[tid * 32 + c]      = f2bf(sv[c] * inv);
            atokS[tid * 32 + 16 + c] = 0;
        }
    }
    __syncthreads();
    // ---- out_tok = a_tok @ v_dp ----
    {
        v16bf a  = lds_fragA(atokS, wave * 16, 32, 0, lane);
        v16bf b0 = lds_fragB(vdpT, 0, 32, 0, lane);
        v16bf b1 = lds_fragB(vdpT, 16, 32, 0, lane);
        v8f d0 = wmma_bf16(a, b0, vz), d1 = wmma_bf16(a, b1, vz);
        float* outp = out_tok + (size_t)blk * 2048;
#pragma unroll
        for (int j = 0; j < 8; ++j) {
            int m = wave * 16 + j + 8 * hi;
            outp[m * 32 + n]      = d0[j];
            outp[m * 32 + 16 + n] = d1[j];
        }
    }
    __syncthreads();
    // ---- s_hid = q^T @ k * sqrt(H); softmax -> a_hid ----
    {
        const int mi = wave >> 1, ni = wave & 1;
        v8f acc = vz;
#pragma unroll
        for (int kk = 0; kk < 64; kk += 32) {
            v16bf a = lds_fragA(qT, mi * 16, 64, kk, lane);
            v16bf b = lds_fragB(kT, ni * 16, 64, kk, lane);
            acc = wmma_bf16(a, b, acc);
        }
#pragma unroll
        for (int j = 0; j < 8; ++j) fS[(mi * 16 + j + 8 * hi) * 32 + ni * 16 + n] = acc[j] * SQRT_H;
    }
    __syncthreads();
    if (tid < 32) {
        float sv[32], mx = -1e30f;
#pragma unroll
        for (int c = 0; c < 32; ++c) { sv[c] = fS[tid * 32 + c]; mx = fmaxf(mx, sv[c]); }
        float s = 0.f;
#pragma unroll
        for (int c = 0; c < 32; ++c) { sv[c] = __expf(sv[c] - mx); s += sv[c]; }
        float inv = 1.0f / s;
#pragma unroll
        for (int c = 0; c < 32; ++c) ahidS[tid * 32 + c] = f2bf(sv[c] * inv);
    }
    __syncthreads();
    // ---- out_hid = v @ a_hid^T ----
    {
        v16bf a  = lds_fragA(vS, wave * 16, 32, 0, lane);
        v16bf b0 = lds_fragB(ahidS, 0, 32, 0, lane);
        v16bf b1 = lds_fragB(ahidS, 16, 32, 0, lane);
        v8f d0 = wmma_bf16(a, b0, vz), d1 = wmma_bf16(a, b1, vz);
        float* outp = out_hid + (size_t)blk * 2048;
#pragma unroll
        for (int j = 0; j < 8; ++j) {
            int m = wave * 16 + j + 8 * hi;
            outp[m * 32 + n]      = d0[j];
            outp[m * 32 + 16 + n] = d1[j];
        }
    }
}

// ---------------------------------------------------------------------------
// Deterministic two-pass BN stats over (65536 x 256) + scale/shift compute.
// ---------------------------------------------------------------------------
__global__ __launch_bounds__(256) void k_stats1(const float* __restrict__ x, float* __restrict__ part) {
    const int c = threadIdx.x, b = blockIdx.x;
    const float* p = x + (size_t)b * 65536 + c;
    float s = 0.f, q = 0.f;
    for (int i = 0; i < 256; ++i) { float v = p[(size_t)i * 256]; s += v; q += v * v; }
    part[b * 256 + c] = s;
    part[65536 + b * 256 + c] = q;
}
__global__ __launch_bounds__(256) void k_stats2(const float* __restrict__ part,
                                                const float* __restrict__ g, const float* __restrict__ bb,
                                                float* __restrict__ scsh) {
    const int c = threadIdx.x;
    float s = 0.f, q = 0.f;
    for (int b = 0; b < 256; ++b) { s += part[b * 256 + c]; q += part[65536 + b * 256 + c]; }
    float mean = s * (1.0f / 65536.0f);
    float var = q * (1.0f / 65536.0f) - mean * mean;
    float sc = g[c] * rsqrtf(var + 1e-5f);
    scsh[c] = sc;
    scsh[256 + c] = bb[c] - mean * sc;
}
__global__ __launch_bounds__(256) void k_bn_apply(float* __restrict__ y, const float* __restrict__ scsh) {
    const int c = threadIdx.x;
    size_t i = (size_t)blockIdx.x * 256 + c;
    y[i] = y[i] * scsh[c] + scsh[256 + c];
}

// ---------------------------------------------------------------------------
extern "C" void kernel_launch(void* const* d_in, const int* in_sizes, int n_in,
                              void* d_out, int out_size, void* d_ws, size_t ws_size,
                              hipStream_t stream) {
    (void)in_sizes; (void)n_in; (void)out_size; (void)ws_size;
    const float* src    = (const float*)d_in[0];
    const float* ema    = (const float*)d_in[1];
    const float* w_qkv  = (const float*)d_in[2];
    const float* b_qkv  = (const float*)d_in[3];
    const float* w_dpk  = (const float*)d_in[4];
    const float* b_dpk  = (const float*)d_in[5];
    const float* w_dpv  = (const float*)d_in[6];
    const float* b_dpv  = (const float*)d_in[7];
    const float* bn1_g  = (const float*)d_in[8];
    const float* bn1_b  = (const float*)d_in[9];
    const float* bn2_g  = (const float*)d_in[10];
    const float* bn2_b  = (const float*)d_in[11];
    const float* bna_g  = (const float*)d_in[12];
    const float* bna_b  = (const float*)d_in[13];
    const float* ff1_w1 = (const float*)d_in[14];
    const float* ff1_b1 = (const float*)d_in[15];
    const float* ff1_w2 = (const float*)d_in[16];
    const float* ff1_b2 = (const float*)d_in[17];
    const float* ff2_w1 = (const float*)d_in[18];
    const float* ff2_b1 = (const float*)d_in[19];
    const float* ff2_w2 = (const float*)d_in[20];
    const float* ff2_b2 = (const float*)d_in[21];
    float* out = (float*)d_out;

    char* w = (char*)d_ws;
    size_t o = 0;
    u16*   qkvB = (u16*)(w + o);   o += (size_t)M_ROWS * 768 * 2;   // 100,663,296
    float* otok = (float*)(w + o); o += (size_t)M_ROWS * 256 * 4;   //  67,108,864
    float* ohid = (float*)(w + o); o += (size_t)M_ROWS * 256 * 4;
    u16*   h1   = (u16*)(w + o);   o += (size_t)M_ROWS * 512 * 2;   //  67,108,864
    u16*   h2   = (u16*)(w + o);   o += (size_t)M_ROWS * 512 * 2;
    u16*   WtQ  = (u16*)(w + o);   o += 768 * 256 * 2;
    u16*   Wt11 = (u16*)(w + o);   o += 512 * 256 * 2;
    u16*   Wt21 = (u16*)(w + o);   o += 512 * 256 * 2;
    u16*   Wt12 = (u16*)(w + o);   o += 256 * 512 * 2;
    u16*   Wt22 = (u16*)(w + o);   o += 256 * 512 * 2;
    u16*   wkT  = (u16*)(w + o);   o += 16 * 32 * 2;
    u16*   wvT  = (u16*)(w + o);   o += 16 * 32 * 2;
    u16*   em64 = (u16*)(w + o);   o += 64 * 64 * 2;
    u16*   em16 = (u16*)(w + o);   o += 16 * 32 * 2;
    float* part = (float*)(w + o); o += 131072 * 4;                 // 524,288
    float* s1   = (float*)(w + o); o += 512 * 4;
    float* s2   = (float*)(w + o); o += 512 * 4;
    float* sa   = (float*)(w + o); o += 512 * 4;

    // 1) weight prep (bf16, N-major)
    k_convert_transpose<<<(256 * 768 + 255) / 256, 256, 0, stream>>>(w_qkv,  WtQ,  256, 768);
    k_convert_transpose<<<(256 * 512 + 255) / 256, 256, 0, stream>>>(ff1_w1, Wt11, 256, 512);
    k_convert_transpose<<<(256 * 512 + 255) / 256, 256, 0, stream>>>(ff2_w1, Wt21, 256, 512);
    k_convert_transpose<<<(512 * 256 + 255) / 256, 256, 0, stream>>>(ff1_w2, Wt12, 512, 256);
    k_convert_transpose<<<(512 * 256 + 255) / 256, 256, 0, stream>>>(ff2_w2, Wt22, 512, 256);
    k_prep_small<<<22, 256, 0, stream>>>(w_dpk, w_dpv, ema, wkT, wvT, em64, em16);

    // 2) QKV projection GEMM (bf16 out)
    k_gemm_a32<<<dim3(M_ROWS / 128, 6), 256, 0, stream>>>(src, WtQ, b_qkv, nullptr, qkvB, 256, 768, 0);

    // 3) dual attention, one head per block
    k_attn<<<8192, 128, 0, stream>>>(qkvB, wkT, wvT, em64, em16, b_dpk, b_dpv, otok, ohid);

    // 4) BN stats for o1 / o2 (deterministic two-pass)
    k_stats1<<<256, 256, 0, stream>>>(otok, part);
    k_stats2<<<1, 256, 0, stream>>>(part, bn1_g, bn1_b, s1);
    k_stats1<<<256, 256, 0, stream>>>(ohid, part);
    k_stats2<<<1, 256, 0, stream>>>(part, bn2_g, bn2_b, s2);

    // 5) FF first GEMMs with fused BN affine + exact GELU
    k_gemm_a32<<<dim3(M_ROWS / 128, 4), 256, 0, stream>>>(otok, Wt11, ff1_b1, s1, h1, 256, 512, 1);
    k_gemm_a32<<<dim3(M_ROWS / 128, 4), 256, 0, stream>>>(ohid, Wt21, ff2_b1, s2, h2, 256, 512, 1);

    // 6) FF second GEMMs (summed) + biases + residual -> d_out (fp32)
    k_gemm_ffout<<<dim3(M_ROWS / 128, 2), 256, 0, stream>>>(h1, Wt12, h2, Wt22, ff1_b2, ff2_b2, src, out);

    // 7) final BN on d_out, in place
    k_stats1<<<256, 256, 0, stream>>>(out, part);
    k_stats2<<<1, 256, 0, stream>>>(part, bna_g, bna_b, sa);
    k_bn_apply<<<M_ROWS, 256, 0, stream>>>(out, sa);
}
